// NGP_18562848653517
// MI455X (gfx1250) — compile-verified, matrix operand order checked by reading, not run on hardware
//
#include <hip/hip_runtime.h>

typedef __attribute__((ext_vector_type(16))) _Float16 v16h;
typedef __attribute__((ext_vector_type(8)))  float    v8f;

#define HASH_N   65536
#define HASH_A   2654435761u
#define WAVES    8
#define TPW      8      // 16-point tiles per wave

// round(2^(4 + i/3)) for i=0..15
__device__ constexpr float kRes[16] = {
    16.f, 20.f, 25.f, 32.f, 40.f, 51.f, 64.f, 81.f,
    102.f, 128.f, 161.f, 203.f, 256.f, 323.f, 406.f, 512.f};

__global__ __launch_bounds__(256) void ngp_hash_mlp_kernel(
    const float* __restrict__ x,      // (M, 2)
    const float* __restrict__ hashf,  // (16, 65536, 2)
    const float* __restrict__ w1,     // (32, 32)
    const float* __restrict__ w2,     // (32, 32)
    const float* __restrict__ w3,     // (3, 32)
    float* __restrict__ out)          // (M, 3)
{
    const int lane = threadIdx.x & 31;
    const int wave = threadIdx.x >> 5;
    const int hb   = lane >> 4;    // which K-half this lane holds (A and B are symmetric)
    const int col  = lane & 15;    // neuron row (for A=weights) / point column (for B, C/D)

    // ---- weight tiles as A operands (rows = output neurons), kept in VGPRs ----
    auto loadW = [&](const float* w, int rows, int nbase) -> v16h {
        v16h aw;
        const int n = nbase + col;                        // output-neuron row
        #pragma unroll
        for (int j = 0; j < 8; ++j) {
            const int k = ((j < 4) ? 2 * j : 8 + 2 * j) + 8 * hb;  // input-feature pair
            float v0 = 0.f, v1 = 0.f;
            if (n < rows) { v0 = w[n * 32 + k]; v1 = w[n * 32 + k + 1]; }
            aw[2 * j]     = (_Float16)v0;
            aw[2 * j + 1] = (_Float16)v1;
        }
        return aw;
    };

    const v16h a1lo = loadW(w1, 32, 0);
    const v16h a1hi = loadW(w1, 32, 16);
    const v16h a2lo = loadW(w2, 32, 0);
    const v16h a2hi = loadW(w2, 32, 16);
    const v16h a3   = loadW(w3, 3, 0);   // rows 3..15 zero-padded

    const v8f  zero  = {};
    const v16h hzero = {};

    const int tile0 = (blockIdx.x * WAVES + wave) * TPW;

    #pragma unroll 2
    for (int t = 0; t < TPW; ++t) {
        const int tile = tile0 + t;
        const int pt   = tile * 16 + col;        // this lane's point (column of B/D)
        const float cx = x[2 * pt + 0];
        const float cy = x[2 * pt + 1];

        // ---- multires hash encode: 8 levels per lane, directly in B layout ----
        // B = feats^T : 32 features (K) x 16 points (N)
        v16h b;
        #pragma unroll
        for (int j = 0; j < 8; ++j) {
            const int lbase = (j < 4) ? j : j + 4;           // compile-time
            const int level = lbase + 4 * hb;                // runtime (hb is 0/1)
            const float res   = hb ? kRes[lbase + 4] : kRes[lbase];
            const float scale = hb ? (float)(1 << (lbase + 4)) : (float)(1 << lbase);

            const float xs = cx * res, ys = cy * res;
            const float xf = floorf(xs), yf = floorf(ys);
            const float wx = xs - xf,   wy = ys - yf;
            const int   xi = (int)xf;
            const int   yi = (int)yf;

            const unsigned ay  = HASH_A * (unsigned)yi;
            const unsigned ay1 = ay + HASH_A;
            const unsigned h00 = ((unsigned)xi       ^ ay ) & (HASH_N - 1);
            const unsigned h01 = ((unsigned)xi       ^ ay1) & (HASH_N - 1);
            const unsigned h10 = ((unsigned)(xi + 1) ^ ay ) & (HASH_N - 1);
            const unsigned h11 = ((unsigned)(xi + 1) ^ ay1) & (HASH_N - 1);

            const float2* tbl = (const float2*)hashf + level * HASH_N;
            const float2 f00 = tbl[h00];
            const float2 f01 = tbl[h01];
            const float2 f10 = tbl[h10];
            const float2 f11 = tbl[h11];

            const float w00 = (1.f - wx) * (1.f - wy);
            const float w01 = (1.f - wx) * wy;
            const float w10 = wx * (1.f - wy);
            const float w11 = wx * wy;
            const float fx = (f00.x * w00 + f01.x * w01 + f10.x * w10 + f11.x * w11) * scale;
            const float fy = (f00.y * w00 + f01.y * w01 + f10.y * w10 + f11.y * w11) * scale;
            b[2 * j]     = (_Float16)fx;
            b[2 * j + 1] = (_Float16)fy;
        }

        // ---- layer 1: h1 = W1 x feats^T  (two stacked 16x16 neuron tiles) ----
        v8f c1lo = __builtin_amdgcn_wmma_f32_16x16x32_f16(false, a1lo, false, b, (short)0, zero, false, false);
        v8f c1hi = __builtin_amdgcn_wmma_f32_16x16x32_f16(false, a1hi, false, b, (short)0, zero, false, false);

        // repack: D rows held by this lane == K set this lane supplies as B.
        // Next-layer B = [c1lo[0..7], c1hi[0..7]] (lane-local cvt), relu in packed f16.
        v16h b2;
        #pragma unroll
        for (int i = 0; i < 8; ++i) {
            b2[i]     = (_Float16)c1lo[i];
            b2[i + 8] = (_Float16)c1hi[i];
        }
        b2 = __builtin_elementwise_max(b2, hzero);   // v_pk_max_num_f16 relu

        // ---- layer 2 ----
        v8f c2lo = __builtin_amdgcn_wmma_f32_16x16x32_f16(false, a2lo, false, b2, (short)0, zero, false, false);
        v8f c2hi = __builtin_amdgcn_wmma_f32_16x16x32_f16(false, a2hi, false, b2, (short)0, zero, false, false);

        v16h b3;
        #pragma unroll
        for (int i = 0; i < 8; ++i) {
            b3[i]     = (_Float16)c2lo[i];
            b3[i + 8] = (_Float16)c2hi[i];
        }
        b3 = __builtin_elementwise_max(b3, hzero);

        // ---- layer 3: rows 0..2 live; each point's 3 outputs land in ONE lane ----
        v8f c3 = __builtin_amdgcn_wmma_f32_16x16x32_f16(false, a3, false, b3, (short)0, zero, false, false);

        if (hb == 0) {
            const float s0 = 1.0f / (1.0f + __expf(-c3[0]));
            const float s1 = 1.0f / (1.0f + __expf(-c3[1]));
            const float s2 = 1.0f / (1.0f + __expf(-c3[2]));
            float* o = out + pt * 3;               // contiguous 12B -> global_store_b96
            o[0] = s0;
            o[1] = s1;
            o[2] = s2;
        }
    }
}

extern "C" void kernel_launch(void* const* d_in, const int* in_sizes, int n_in,
                              void* d_out, int out_size, void* d_ws, size_t ws_size,
                              hipStream_t stream) {
    const float* x  = (const float*)d_in[0];
    const float* hf = (const float*)d_in[1];
    const float* w1 = (const float*)d_in[2];
    const float* w2 = (const float*)d_in[3];
    const float* w3 = (const float*)d_in[4];
    float* out = (float*)d_out;

    const int M = in_sizes[0] / 2;                 // 2,097,152
    const int pointsPerBlock = WAVES * TPW * 16;   // 1024
    const int grid = (M + pointsPerBlock - 1) / pointsPerBlock;  // 2048

    ngp_hash_mlp_kernel<<<grid, 256, 0, stream>>>(x, hf, w1, w2, w3, out);
}
